// MultiHeadAttention_91018946936837
// MI455X (gfx1250) — matrix operand order repaired and checked by reading, not compiled, not run on hardware
//
#include <hip/hip_runtime.h>
#include <hip/hip_bf16.h>

typedef _Float16 v4h  __attribute__((ext_vector_type(4)));
typedef _Float16 v8h  __attribute__((ext_vector_type(8)));
typedef _Float16 v16h __attribute__((ext_vector_type(16)));
typedef float    v8f  __attribute__((ext_vector_type(8)));

#define DEV __device__ __forceinline__

// ---------------------------------------------------------------------------
// WMMA f32 = f16 x f16 + f32   (16x16x32)
// ---------------------------------------------------------------------------
DEV v8f wmma_f16(v16h a, v16h b, v8f c) {
  return __builtin_amdgcn_wmma_f32_16x16x32_f16(
      /*neg_a=*/false, a, /*neg_b=*/false, b,
      /*c_mod=*/(short)0, c, /*reuse_a=*/false, /*reuse_b=*/false);
}

// A-matrix fragment loader: A is 16x32 f16, row-major with leading dim ld.
// ISA layout (16-bit A 16x32): lanes 0-15 row M=lane, VGPR0..3 = K 0..7,
// VGPR4..7 = K 16..23; lanes 16-31 same rows, K 8..15 and K 24..31.
DEV v16h load_fragA(const _Float16* base, int ld) {
  const int lane = threadIdx.x & 31;
  const int r    = lane & 15;
  const int klo  = (lane >> 4) << 3;                 // 0 or 8
  const _Float16* p = base + (size_t)r * ld + klo;
  v8h lo = *reinterpret_cast<const v8h*>(p);         // K = klo .. klo+7
  v8h hi = *reinterpret_cast<const v8h*>(p + 16);    // K = klo+16 .. klo+23
  v16h f;
#pragma unroll
  for (int i = 0; i < 8; ++i) { f[i] = lo[i]; f[i + 8] = hi[i]; }
  return f;
}

// B-matrix fragment loader: caller provides B^T (16 rows = N, 32 cols = K),
// row-major with leading dim ld. Layout matches load_fragA: lanes 0-15
// column N=lane hold K 0..7, 16..23; lanes 16-31 hold K 8..15, 24..31.
DEV v16h load_fragB(const _Float16* baseT, int ld) {
  const int lane = threadIdx.x & 31;
  const int n    = lane & 15;
  const int ks   = (lane >> 4) << 3;                 // 0 or 8
  const _Float16* p = baseT + (size_t)n * ld + ks;
  v8h lo = *reinterpret_cast<const v8h*>(p);         // K = ks .. ks+7
  v8h hi = *reinterpret_cast<const v8h*>(p + 16);    // K = ks+16 .. ks+23
  v16h f;
#pragma unroll
  for (int i = 0; i < 8; ++i) { f[i] = lo[i]; f[i + 8] = hi[i]; }
  return f;
}

// ---------------------------------------------------------------------------
// Elementwise conversion kernels
// ---------------------------------------------------------------------------
__global__ __launch_bounds__(256) void cvt_x_kernel(const float* __restrict__ x,
                                                    _Float16* __restrict__ x16,
                                                    int n) {
  int i = (blockIdx.x * 256 + threadIdx.x) * 4;
  if (i < n) {
    float4 f = *reinterpret_cast<const float4*>(x + i);
    v4h h; h[0] = (_Float16)f.x; h[1] = (_Float16)f.y;
           h[2] = (_Float16)f.z; h[3] = (_Float16)f.w;
    *reinterpret_cast<v4h*>(x16 + i) = h;
  }
}

// LDS-tiled transpose + f32->f16 for the four 1024x1024 weight matrices.
// Output wT[m][n*1024 + i] = W_m[i][n]. Coalesced reads AND writes.
__global__ __launch_bounds__(256) void cvt_w_kernel(const float* __restrict__ w0,
                                                    const float* __restrict__ w1,
                                                    const float* __restrict__ w2,
                                                    const float* __restrict__ w3,
                                                    _Float16* __restrict__ wT) {
  const int D = 1024;
  const int m = blockIdx.z;
  const float* w = (m == 0) ? w0 : (m == 1) ? w1 : (m == 2) ? w2 : w3;
  __shared__ float tile[32][33];                     // pad -> no bank conflicts
  const int i0 = blockIdx.x * 32;                    // input-row block
  const int n0 = blockIdx.y * 32;                    // input-col block
  const int c  = threadIdx.x & 31;
  const int r8 = threadIdx.x >> 5;                   // 0..7
#pragma unroll
  for (int p = 0; p < 4; ++p) {
    const int r = r8 + p * 8;
    tile[r][c] = w[(size_t)(i0 + r) * D + n0 + c];
  }
  __syncthreads();
#pragma unroll
  for (int p = 0; p < 4; ++p) {
    const int r = r8 + p * 8;
    wT[(size_t)m * D * D + (size_t)(n0 + r) * D + i0 + c] = (_Float16)tile[c][r];
  }
}

// ---------------------------------------------------------------------------
// QKV projection GEMM: C(M x 1024) = x16(M x 1024) * W + bias
// One wave computes a 64x64 tile (4x4 WMMA tiles): 16 b128 loads per 16 wmma.
// Block = 8 waves = 128(m) x 256(n).
// proj 0/1 -> Q/K stored as (B,H,S,HD) f16; proj 2 -> V stored as (B,H,HD,S).
// ---------------------------------------------------------------------------
__global__ __launch_bounds__(256) void gemm_qkv_kernel(
    const _Float16* __restrict__ x16, const _Float16* __restrict__ wT,
    const float* __restrict__ bq, const float* __restrict__ bk,
    const float* __restrict__ bv,
    _Float16* __restrict__ qO, _Float16* __restrict__ kO,
    _Float16* __restrict__ vtO) {
  const int D = 1024, S = 2048, H = 16, HD = 64;
  const int proj = blockIdx.z;
  const _Float16* w = wT + (size_t)proj * D * D;
  const float* bias = (proj == 0) ? bq : (proj == 1) ? bk : bv;
  const int wave = threadIdx.x >> 5;
  const int lane = threadIdx.x & 31;
  const int m0 = blockIdx.x * 128 + (wave & 1) * 64;
  const int n0 = blockIdx.y * 256 + (wave >> 1) * 64;

  v8f c[4][4];
#pragma unroll
  for (int i = 0; i < 4; ++i)
#pragma unroll
    for (int j = 0; j < 4; ++j) c[i][j] = {};

  for (int k0 = 0; k0 < D; k0 += 32) {
    v16h a[4], bb[4];
#pragma unroll
    for (int i = 0; i < 4; ++i)
      a[i] = load_fragA(x16 + (size_t)(m0 + i * 16) * D + k0, D);
#pragma unroll
    for (int j = 0; j < 4; ++j)
      bb[j] = load_fragB(w + (size_t)(n0 + j * 16) * D + k0, D);
#pragma unroll
    for (int i = 0; i < 4; ++i)
#pragma unroll
      for (int j = 0; j < 4; ++j) c[i][j] = wmma_f16(a[i], bb[j], c[i][j]);
  }

  const int lh = lane >> 4, nl = lane & 15;
#pragma unroll
  for (int j = 0; j < 4; ++j) {
    const int gn = n0 + j * 16 + nl;
    const float bsc = bias[gn];
    const int h = gn >> 6, hd = gn & 63;
#pragma unroll
    for (int i = 0; i < 4; ++i) {
#pragma unroll
      for (int v = 0; v < 8; ++v) {
        const int gm = m0 + i * 16 + v + 8 * lh;
        const int bI = gm >> 11;                     // S = 2048
        const int s  = gm & 2047;
        const float val = c[i][j][v] + bsc;
        const size_t bh = (size_t)(bI * H + h);
        if (proj == 0)      qO[(bh * S + s) * HD + hd] = (_Float16)val;
        else if (proj == 1) kO[(bh * S + s) * HD + hd] = (_Float16)val;
        else                vtO[(bh * HD + hd) * S + s] = (_Float16)val;
      }
    }
  }
}

// ---------------------------------------------------------------------------
// Flash attention: one wave per (b, h, 16-query tile). Causal + key mask.
// ---------------------------------------------------------------------------
__global__ __launch_bounds__(256) void attn_kernel(
    const _Float16* __restrict__ Q, const _Float16* __restrict__ K,
    const _Float16* __restrict__ Vt, const int* __restrict__ amask,
    _Float16* __restrict__ attn) {
  const int S = 2048, H = 16, HD = 64, D = 1024;
  __shared__ __align__(16) _Float16 pbuf_all[8][16 * 32];
  const int wave = threadIdx.x >> 5;
  const int lane = threadIdx.x & 31;
  _Float16* pbuf = &pbuf_all[wave][0];

  const int wg = blockIdx.x * 8 + wave;
  const int qb = wg & 127;                           // S/16 = 128
  const int h  = (wg >> 7) & 15;
  const int b  = wg >> 11;
  const int qs = qb * 16;

  const _Float16* Qb = Q  + (size_t)(b * H + h) * S * HD;
  const _Float16* Kb = K  + (size_t)(b * H + h) * S * HD;
  const _Float16* Vb = Vt + (size_t)(b * H + h) * HD * S;
  const int* mk = amask + b * S;

  const v16h aq0 = load_fragA(Qb + (size_t)qs * HD + 0, HD);
  const v16h aq1 = load_fragA(Qb + (size_t)qs * HD + 32, HD);

  v8f o[4];
#pragma unroll
  for (int t = 0; t < 4; ++t) o[t] = {};
  const float NEG_INF = -__builtin_inff();
  float mrow[8], lrow[8];
#pragma unroll
  for (int v = 0; v < 8; ++v) { mrow[v] = NEG_INF; lrow[v] = 0.f; }

  const int lh = lane >> 4, nl = lane & 15;
  const float scale = 0.125f;                        // 1/sqrt(64)

  for (int kb = 0; kb < qs + 16; kb += 32) {
    // scores: 16 queries x 32 keys  (two 16x16 tiles)
    v8f s0 = {}, s1 = {};
    s0 = wmma_f16(aq0, load_fragB(Kb + (size_t)kb * HD + 0, HD), s0);
    s0 = wmma_f16(aq1, load_fragB(Kb + (size_t)kb * HD + 32, HD), s0);
    s1 = wmma_f16(aq0, load_fragB(Kb + (size_t)(kb + 16) * HD + 0, HD), s1);
    s1 = wmma_f16(aq1, load_fragB(Kb + (size_t)(kb + 16) * HD + 32, HD), s1);

    const int key0 = kb + nl, key1 = kb + 16 + nl;
    const bool mv0 = mk[key0] != 0, mv1 = mk[key1] != 0;

#pragma unroll
    for (int v = 0; v < 8; ++v) {
      const int q = qs + v + 8 * lh;
      float x0 = (mv0 && key0 <= q) ? s0[v] * scale : NEG_INF;
      float x1 = (mv1 && key1 <= q) ? s1[v] * scale : NEG_INF;
      // block row-max across the 16 lanes holding this row
      float bm = fmaxf(x0, x1);
      bm = fmaxf(bm, __shfl_xor(bm, 1, 32));
      bm = fmaxf(bm, __shfl_xor(bm, 2, 32));
      bm = fmaxf(bm, __shfl_xor(bm, 4, 32));
      bm = fmaxf(bm, __shfl_xor(bm, 8, 32));
      const float newm  = fmaxf(mrow[v], bm);
      const float alpha = (mrow[v] == NEG_INF) ? 0.f : __expf(mrow[v] - newm);
      float p0 = (newm == NEG_INF) ? 0.f : __expf(x0 - newm);
      float p1 = (newm == NEG_INF) ? 0.f : __expf(x1 - newm);
      float rs = p0 + p1;
      rs += __shfl_xor(rs, 1, 32);
      rs += __shfl_xor(rs, 2, 32);
      rs += __shfl_xor(rs, 4, 32);
      rs += __shfl_xor(rs, 8, 32);
      lrow[v] = lrow[v] * alpha + rs;
      mrow[v] = newm;
#pragma unroll
      for (int t = 0; t < 4; ++t) o[t][v] *= alpha;
      const int r = v + 8 * lh;
      pbuf[r * 32 + nl]      = (_Float16)p0;         // D-layout -> row-major
      pbuf[r * 32 + 16 + nl] = (_Float16)p1;
    }
    asm volatile("s_wait_dscnt 0" ::: "memory");     // LDS writes visible
    const v16h ap = load_fragA(pbuf, 32);            // re-read in A layout
#pragma unroll
    for (int t = 0; t < 4; ++t)
      o[t] = wmma_f16(ap, load_fragB(Vb + (size_t)(t * 16) * S + kb, S), o[t]);
  }

#pragma unroll
  for (int v = 0; v < 8; ++v) {
    const float inv = (lrow[v] > 0.f) ? 1.f / lrow[v] : 0.f;  // nan_to_num
    const int s = qs + v + 8 * lh;
#pragma unroll
    for (int t = 0; t < 4; ++t)
      attn[(size_t)(b * S + s) * D + h * HD + t * 16 + nl] =
          (_Float16)(o[t][v] * inv);
  }
}

// ---------------------------------------------------------------------------
// Output projection: out(M x 1024, f32) = attn16 * Wp + bp
// Same 64x64-per-wave tiling as the QKV GEMM.
// ---------------------------------------------------------------------------
__global__ __launch_bounds__(256) void gemm_out_kernel(
    const _Float16* __restrict__ a16, const _Float16* __restrict__ wT,
    const float* __restrict__ bias, float* __restrict__ out) {
  const int D = 1024;
  const int wave = threadIdx.x >> 5;
  const int lane = threadIdx.x & 31;
  const int m0 = blockIdx.x * 128 + (wave & 1) * 64;
  const int n0 = blockIdx.y * 256 + (wave >> 1) * 64;

  v8f c[4][4];
#pragma unroll
  for (int i = 0; i < 4; ++i)
#pragma unroll
    for (int j = 0; j < 4; ++j) c[i][j] = {};

  for (int k0 = 0; k0 < D; k0 += 32) {
    v16h a[4], bb[4];
#pragma unroll
    for (int i = 0; i < 4; ++i)
      a[i] = load_fragA(a16 + (size_t)(m0 + i * 16) * D + k0, D);
#pragma unroll
    for (int j = 0; j < 4; ++j)
      bb[j] = load_fragB(wT + (size_t)(n0 + j * 16) * D + k0, D);
#pragma unroll
    for (int i = 0; i < 4; ++i)
#pragma unroll
      for (int j = 0; j < 4; ++j) c[i][j] = wmma_f16(a[i], bb[j], c[i][j]);
  }

  const int lh = lane >> 4, nl = lane & 15;
#pragma unroll
  for (int j = 0; j < 4; ++j) {
    const int gn = n0 + j * 16 + nl;
    const float bsc = bias[gn];
#pragma unroll
    for (int i = 0; i < 4; ++i)
#pragma unroll
      for (int v = 0; v < 8; ++v) {
        const int gm = m0 + i * 16 + v + 8 * lh;
        out[(size_t)gm * D + gn] = c[i][j][v] + bsc;
      }
  }
}

// ---------------------------------------------------------------------------
// Host launcher
// ---------------------------------------------------------------------------
extern "C" void kernel_launch(void* const* d_in, const int* in_sizes, int n_in,
                              void* d_out, int out_size, void* d_ws,
                              size_t ws_size, hipStream_t stream) {
  (void)n_in; (void)out_size; (void)ws_size;
  const int S = 2048, D = 1024, H = 16;
  const float* x   = (const float*)d_in[0];
  const int* amask = (const int*)d_in[1];
  const float* Wq  = (const float*)d_in[2];
  const float* bq  = (const float*)d_in[3];
  const float* Wk  = (const float*)d_in[4];
  const float* bk  = (const float*)d_in[5];
  const float* Wv  = (const float*)d_in[6];
  const float* bv  = (const float*)d_in[7];
  const float* Wp  = (const float*)d_in[8];
  const float* bp  = (const float*)d_in[9];
  float* out = (float*)d_out;

  const int B = in_sizes[0] / (S * D);
  const int M = B * S;

  // workspace layout (72 MB for B=4); x16 region is reused for attn output
  char* ws = (char*)d_ws;
  size_t off = 0;
  _Float16* x16 = (_Float16*)(ws + off); off += (size_t)M * D * 2;
  _Float16* wT  = (_Float16*)(ws + off); off += (size_t)4 * D * D * 2;
  _Float16* q16 = (_Float16*)(ws + off); off += (size_t)M * D * 2;
  _Float16* k16 = (_Float16*)(ws + off); off += (size_t)M * D * 2;
  _Float16* vt16= (_Float16*)(ws + off); off += (size_t)M * D * 2;
  _Float16* attn16 = x16;   // alias: x16 dead after QKV projections

  const int nX = M * D;
  cvt_x_kernel<<<dim3((nX / 4 + 255) / 256), 256, 0, stream>>>(x, x16, nX);
  cvt_w_kernel<<<dim3(D / 32, D / 32, 4), 256, 0, stream>>>(Wq, Wk, Wv, Wp, wT);
  gemm_qkv_kernel<<<dim3(M / 128, D / 256, 3), 256, 0, stream>>>(
      x16, wT, bq, bk, bv, q16, k16, vt16);
  attn_kernel<<<dim3(B * H * (S / 16) / 8), 256, 0, stream>>>(
      q16, k16, vt16, amask, attn16);
  gemm_out_kernel<<<dim3(M / 128, D / 256), 256, 0, stream>>>(
      attn16, wT + (size_t)3 * D * D, bp, out);
}
